// StateSpaceAttention_21466246545688
// MI455X (gfx1250) — matrix operand order repaired and checked by reading, not compiled, not run on hardware
//
#include <hip/hip_runtime.h>
#include <stdint.h>

#define T_LEN 2048
#define NHEAD 16
#define HDIM  64
#define EMB   1024
#define QKV3  3072

typedef __attribute__((ext_vector_type(16))) __bf16 v16bf;
typedef __attribute__((ext_vector_type(8)))  float  v8f;
typedef __attribute__((ext_vector_type(4)))  unsigned int u32x4;
typedef __attribute__((ext_vector_type(8)))  int          i32x8;
typedef __attribute__((ext_vector_type(4)))  int          i32x4;

__device__ __forceinline__ unsigned short f2bf(float f) {
  union { float f; uint32_t u; } x; x.f = f;
  uint32_t r = x.u + 0x7FFFu + ((x.u >> 16) & 1u);
  return (unsigned short)(r >> 16);
}

__global__ __launch_bounds__(256) void cvt_f32_to_bf16(
    const float* __restrict__ in, unsigned short* __restrict__ out, int n) {
  int i = blockIdx.x * blockDim.x + threadIdx.x;
  if (i < n) out[i] = f2bf(in[i]);
}

// ---------------------------------------------------------------------------
// Pack B[K,N] (f32, row-major) into WMMA B-fragment order (bf16):
// out[(((nb*(K/32) + kc)*32 + lane)*16 + e] = bf16(B[kc*32 + kb16 + e][nb*16 + (lane&15)])
// where kb16 = (lane>=16)?16:0.  One 16-element run (32B) per thread.
// ---------------------------------------------------------------------------
__global__ __launch_bounds__(256) void pack_b_bf16(
    const float* __restrict__ B, unsigned short* __restrict__ out, int K, int N) {
  int idx = blockIdx.x * 256 + threadIdx.x;
  int runs = (N / 16) * (K / 32) * 32;
  if (idx >= runs) return;
  int lane = idx & 31;
  int kc   = (idx >> 5) % (K / 32);
  int nb   = (idx >> 5) / (K / 32);
  int n    = nb * 16 + (lane & 15);
  int kb   = kc * 32 + ((lane >= 16) ? 16 : 0);
  union { uint4 u[2]; unsigned short s[16]; } t;
#pragma unroll
  for (int e = 0; e < 16; ++e) t.s[e] = f2bf(B[(size_t)(kb + e) * N + n]);
  uint4* op = (uint4*)&out[(size_t)idx * 16];
  op[0] = t.u[0]; op[1] = t.u[1];
}

// ---------------------------------------------------------------------------
// bf16 WMMA GEMM: C[M,N](f32) = A[M,K](bf16 row-major) * Bp (pre-packed frags).
// Block 128x128, BK=32, 256 threads = 8 waves (2x4), wave tile 64x32.
// A tile double-buffered in LDS via global_load_async_to_lds_b128;
// B fragments streamed directly from global (L2-resident packed layout).
// ---------------------------------------------------------------------------
__global__ __launch_bounds__(256) void gemm_bf16_wmma(
    const unsigned short* __restrict__ A, const unsigned short* __restrict__ Bp,
    float* __restrict__ C, int M, int N, int K) {
  __shared__ unsigned short As[2][128][32];   // [buf][m][k]
  const int tid  = threadIdx.x;
  const int lane = tid & 31;
  const int wave = tid >> 5;
  const int wm   = wave >> 2;     // 0..1
  const int wn   = wave & 3;      // 0..3
  const int bm   = blockIdx.y * 128;
  const int bn   = blockIdx.x * 128;
  const int l15  = lane & 15;
  const int kb8  = (lane >= 16) ? 8 : 0;
  const int K32  = K / 32;

  const v8f vzero = {};
  v8f acc[4][2];
#pragma unroll
  for (int i = 0; i < 4; ++i)
#pragma unroll
    for (int j = 0; j < 2; ++j) acc[i][j] = vzero;

  // A staging: thread covers 16 halfwords (2 x 16B async copies)
  const int arow = tid >> 1;
  const int acol = (tid & 1) * 16;
  const unsigned ldsA[2] = {
    (unsigned)(uintptr_t)&As[0][arow][acol],
    (unsigned)(uintptr_t)&As[1][arow][acol]
  };
  unsigned goffA = (unsigned)(((size_t)(bm + arow) * (size_t)K + (size_t)acol) * 2u);
  const unsigned long long abase = (unsigned long long)(uintptr_t)A;

  // B fragment stream pointers (packed layout)
  const unsigned short* bp0 =
      Bp + ((size_t)((bn >> 4) + wn * 2 + 0) * K32 * 32 + lane) * 16;
  const unsigned short* bp1 = bp0 + (size_t)K32 * 512;

  union F { uint4 u[2]; v16bf v; };

  // prologue: stage tile 0 into buffer 0
  asm volatile("global_load_async_to_lds_b128 %0, %1, %2"
               :: "v"(ldsA[0]), "v"(goffA), "s"(abase) : "memory");
  asm volatile("global_load_async_to_lds_b128 %0, %1, %2"
               :: "v"(ldsA[0] + 16u), "v"(goffA + 16u), "s"(abase) : "memory");
  goffA += 64u;

  for (int i = 0; i < K32; ++i) {
    const int cur = i & 1;
    asm volatile("s_wait_asynccnt 0" ::: "memory");
    __syncthreads();                       // As[cur] visible to all waves
    if (i + 1 < K32) {                     // overlap: stage next tile now
      asm volatile("global_load_async_to_lds_b128 %0, %1, %2"
                   :: "v"(ldsA[cur ^ 1]), "v"(goffA), "s"(abase) : "memory");
      asm volatile("global_load_async_to_lds_b128 %0, %1, %2"
                   :: "v"(ldsA[cur ^ 1] + 16u), "v"(goffA + 16u), "s"(abase) : "memory");
      goffA += 64u;
      __builtin_prefetch(bp0 + 512, 0, 1);
    }
    // B fragments straight from global (2 x 16B contiguous per lane)
    F b0, b1;
    {
      const uint4* p0 = (const uint4*)bp0;
      const uint4* p1 = (const uint4*)bp1;
      b0.u[0] = p0[0]; b0.u[1] = p0[1];
      b1.u[0] = p1[0]; b1.u[1] = p1[1];
      bp0 += 512; bp1 += 512;
    }
    // A fragments from LDS
#pragma unroll
    for (int mi = 0; mi < 4; ++mi) {
      F a;
      int r = wm * 64 + mi * 16 + l15;
      a.u[0] = *(const uint4*)&As[cur][r][2 * kb8];
      a.u[1] = *(const uint4*)&As[cur][r][2 * kb8 + 8];
      acc[mi][0] = __builtin_amdgcn_wmma_f32_16x16x32_bf16(
          false, a.v, false, b0.v, (short)0, acc[mi][0], false, false);
      acc[mi][1] = __builtin_amdgcn_wmma_f32_16x16x32_bf16(
          false, a.v, false, b1.v, (short)0, acc[mi][1], false, false);
    }
  }

#pragma unroll
  for (int mi = 0; mi < 4; ++mi)
#pragma unroll
    for (int ni = 0; ni < 2; ++ni) {
      int col   = bn + wn * 32 + ni * 16 + l15;
      int rbase = bm + wm * 64 + mi * 16 + ((lane >= 16) ? 8 : 0);
#pragma unroll
      for (int r = 0; r < 8; ++r)
        C[(size_t)(rbase + r) * N + col] = acc[mi][ni][r];
    }
}

// ---------------------------------------------------------------------------
// Fused depthwise causal conv1d(4) + SiLU + RoPE.
// ---------------------------------------------------------------------------
__global__ __launch_bounds__(512) void conv_silu_rope(
    const float* __restrict__ qkv, const float* __restrict__ qw,
    const float* __restrict__ kw, const float* __restrict__ vw,
    const float* __restrict__ cosb, const float* __restrict__ sinb,
    float* __restrict__ qf, float* __restrict__ kf, float* __restrict__ vf,
    unsigned short* __restrict__ qb, unsigned short* __restrict__ kb) {
  const int t  = blockIdx.x;
  const int h  = threadIdx.x >> 5;
  const int dh = threadIdx.x & 31;
  const float c = cosb[t * 32 + dh];
  const float s = sinb[t * 32 + dh];
  float y[3][2];
#pragma unroll
  for (int sg = 0; sg < 3; ++sg) {
    const float* w = (sg == 0) ? qw : (sg == 1) ? kw : vw;
#pragma unroll
    for (int half = 0; half < 2; ++half) {
      int d   = dh + half * 32;
      int ch  = h * 64 + d;
      int src = h * 192 + sg * 64 + d;
      float acc = 0.f;
#pragma unroll
      for (int tau = 0; tau < 4; ++tau) {
        int ts = t - 3 + tau;
        float xv = (ts >= 0) ? qkv[(size_t)ts * QKV3 + src] : 0.f;
        acc += xv * w[ch * 4 + tau];
      }
      y[sg][half] = acc / (1.f + __expf(-acc));   // SiLU
    }
  }
  size_t o1 = (size_t)t * EMB + h * 64 + dh;
  size_t o2 = o1 + 32;
  float q1 = y[0][0] * c - y[0][1] * s, q2 = y[0][1] * c + y[0][0] * s;
  float k1 = y[1][0] * c - y[1][1] * s, k2 = y[1][1] * c + y[1][0] * s;
  qf[o1] = q1; qf[o2] = q2;
  kf[o1] = k1; kf[o2] = k2;
  vf[o1] = y[2][0]; vf[o2] = y[2][1];
  qb[o1] = f2bf(q1); qb[o2] = f2bf(q2);
  kb[o1] = f2bf(k1); kb[o2] = f2bf(k2);
}

// ---------------------------------------------------------------------------
// Gated delta rule: sequential over T, one block per head, S[64][64] in LDS.
// ---------------------------------------------------------------------------
__global__ __launch_bounds__(256) void delta_scan(
    const float* __restrict__ qf, const float* __restrict__ kf,
    const float* __restrict__ vf, const float* __restrict__ beta,
    const float* __restrict__ g, unsigned short* __restrict__ vob) {
  __shared__ float S[64][64];
  __shared__ float kv[64], qv[64], vv[64], dl[64];
  __shared__ float part[4][64];
  __shared__ float sc[2];
  const int tid = threadIdx.x;
  const int col = tid & 63;
  const int kp  = tid >> 6;
  const int h   = blockIdx.x;
  for (int i = tid; i < 64 * 64; i += 256) (&S[0][0])[i] = 0.f;
  __syncthreads();
  const float scale = 0.125f;
  for (int t = 0; t < T_LEN; ++t) {
    if (tid < 64) {
      size_t off = (size_t)t * EMB + h * 64 + tid;
      kv[tid] = kf[off]; qv[tid] = qf[off]; vv[tid] = vf[off];
    }
    if (tid == 0) { sc[0] = beta[t * NHEAD + h]; sc[1] = g[t * NHEAD + h]; }
    __syncthreads();
    float gg = sc[1], accp = 0.f;
#pragma unroll
    for (int i = 0; i < 16; ++i) {
      int kk = kp * 16 + i;
      float sv = S[kk][col] * gg;
      S[kk][col] = sv;
      accp += kv[kk] * sv;
    }
    part[kp][col] = accp;
    __syncthreads();
    if (tid < 64) {
      float pred = part[0][tid] + part[1][tid] + part[2][tid] + part[3][tid];
      dl[tid] = sc[0] * (vv[tid] - pred);
    }
    __syncthreads();
    float dd = dl[col], acco = 0.f;
#pragma unroll
    for (int i = 0; i < 16; ++i) {
      int kk = kp * 16 + i;
      float sv = S[kk][col] + kv[kk] * dd;
      S[kk][col] = sv;
      acco += qv[kk] * sv;
    }
    part[kp][col] = acco;
    __syncthreads();
    if (tid < 64)
      vob[(size_t)t * EMB + h * 64 + tid] =
          f2bf(scale * (part[0][tid] + part[1][tid] + part[2][tid] + part[3][tid]));
    __syncthreads();
  }
}

// ---------------------------------------------------------------------------
// Causal flash attention with WMMA; K tile staged by the Tensor Data Mover.
// Grid (32 q-blocks, 16 heads), 128 threads = 4 waves.
// ---------------------------------------------------------------------------
__global__ __launch_bounds__(128) void flash_attn(
    const unsigned short* __restrict__ qb, const unsigned short* __restrict__ kb,
    const unsigned short* __restrict__ vb, unsigned short* __restrict__ yb) {
  __shared__ unsigned short Ks[64][64];   // [kvrow][d]  (filled by TDM)
  __shared__ unsigned short Vt[64][64];   // transposed: [d][kvrow]
  __shared__ float          Sb[64][64];
  __shared__ unsigned short Pb[64][64];
  __shared__ float          alpha[64];
  const int tid  = threadIdx.x;
  const int lane = tid & 31;
  const int wave = tid >> 5;
  const int h    = blockIdx.y;
  const int q0   = blockIdx.x * 64;
  const int l15  = lane & 15;
  const int kb8  = (lane >= 16) ? 8 : 0;
  const int kb16 = (lane >= 16) ? 16 : 0;
  const float scale = 0.125f;

  union F { uint4 u[2]; v16bf v; };
  F qa[2];
  {
    const unsigned short* qrow = qb + (size_t)(q0 + wave * 16 + l15) * EMB + h * 64;
#pragma unroll
    for (int cb = 0; cb < 2; ++cb) {
      qa[cb].u[0] = *(const uint4*)(qrow + cb * 32 + kb16);
      qa[cb].u[1] = *(const uint4*)(qrow + cb * 32 + kb16 + 8);
    }
  }
  const v8f vzero = {};
  v8f of[4];
#pragma unroll
  for (int f = 0; f < 4; ++f) of[f] = vzero;
  float mrow = -1e30f, lrow = 0.f;   // valid for tid<64 only

  const unsigned ldsKs = (unsigned)(uintptr_t)&Ks[0][0];
  const unsigned long long kbase = (unsigned long long)(uintptr_t)kb;

  for (int jb = 0; jb <= (int)blockIdx.x; ++jb) {
    // ---- K tile via Tensor Data Mover: 64x64 bf16, row stride EMB ----
    if (wave == 0) {
      unsigned long long gaddr =
          kbase + (((unsigned long long)(jb * 64) * EMB + h * 64) << 1);
      u32x4 g0 = { 1u,                      // count=1, user desc
                   ldsKs,                   // LDS destination
                   (unsigned)gaddr,
                   ((unsigned)(gaddr >> 32) & 0x01FFFFFFu) | 0x80000000u }; // type=2
      i32x8 g1 = { (int)(1u << 16),         // data_size = 2 bytes
                   (int)(64u << 16),        // tensor_dim0 = 64 (lo16 @ bits 63:48)
                   (int)(64u << 16),        // tensor_dim1 lo16
                   (int)(64u << 16),        // tile_dim0 = 64
                   64,                      // tile_dim1 = 64 (tile_dim2 = 0 -> 2D)
                   (int)EMB,                // tensor_dim0_stride = 1024
                   0, 0 };
      i32x4 z4 = { 0, 0, 0, 0 };
      i32x8 z8 = { 0, 0, 0, 0, 0, 0, 0, 0 };
      __builtin_amdgcn_tensor_load_to_lds(g0, g1, z4, z4, z8, 0);
    }
    // ---- V tile staged transposed (manual, all threads) ----
    {
      int r  = tid >> 1;
      int cb = (tid & 1) * 32;
      union { uint4 u[4]; unsigned short s[32]; } tmp;
      const uint4* vr = (const uint4*)(vb + (size_t)(jb * 64 + r) * EMB + h * 64 + cb);
      tmp.u[0] = vr[0]; tmp.u[1] = vr[1]; tmp.u[2] = vr[2]; tmp.u[3] = vr[3];
#pragma unroll
      for (int j = 0; j < 32; ++j) Vt[cb + j][r] = tmp.s[j];
    }
    if (wave == 0) __builtin_amdgcn_s_wait_tensorcnt(0);
    __syncthreads();

    // S = Q * K^T
#pragma unroll
    for (int f = 0; f < 4; ++f) {
      v8f s = vzero;
#pragma unroll
      for (int cb = 0; cb < 2; ++cb) {
        F b;
        int n = f * 16 + l15;
        b.u[0] = *(const uint4*)&Ks[n][cb * 32 + kb16];
        b.u[1] = *(const uint4*)&Ks[n][cb * 32 + kb16 + 8];
        s = __builtin_amdgcn_wmma_f32_16x16x32_bf16(
                false, qa[cb].v, false, b.v, (short)0, s, false, false);
      }
      int rb = wave * 16 + ((lane >= 16) ? 8 : 0);
#pragma unroll
      for (int r = 0; r < 8; ++r) Sb[rb + r][f * 16 + l15] = s[r];
    }
    __syncthreads();

    // online softmax, one thread per row
    if (tid < 64) {
      int row    = tid;
      int nvalid = (jb == (int)blockIdx.x) ? (row + 1) : 64;
      float bm = -1e30f;
      for (int c2 = 0; c2 < nvalid; ++c2) bm = fmaxf(bm, Sb[row][c2] * scale);
      float nm = fmaxf(mrow, bm);
      float al = __expf(mrow - nm);
      float ssum = 0.f;
      for (int c2 = 0; c2 < 64; ++c2) {
        float p = (c2 < nvalid) ? __expf(Sb[row][c2] * scale - nm) : 0.f;
        ssum += p;
        Pb[row][c2] = f2bf(p);
      }
      lrow = lrow * al + ssum;
      mrow = nm;
      alpha[row] = al;
    }
    __syncthreads();

    // O = O*alpha + P*V
    {
      int rb = wave * 16 + ((lane >= 16) ? 8 : 0);
#pragma unroll
      for (int f = 0; f < 4; ++f)
#pragma unroll
        for (int r = 0; r < 8; ++r) of[f][r] *= alpha[rb + r];
    }
#pragma unroll
    for (int cb = 0; cb < 2; ++cb) {
      F pa;
      int prow = wave * 16 + l15;
      pa.u[0] = *(const uint4*)&Pb[prow][cb * 32 + kb16];
      pa.u[1] = *(const uint4*)&Pb[prow][cb * 32 + kb16 + 8];
#pragma unroll
      for (int f = 0; f < 4; ++f) {
        F vfr;
        int n = f * 16 + l15;
        vfr.u[0] = *(const uint4*)&Vt[n][cb * 32 + kb16];
        vfr.u[1] = *(const uint4*)&Vt[n][cb * 32 + kb16 + 8];
        of[f] = __builtin_amdgcn_wmma_f32_16x16x32_bf16(
                    false, pa.v, false, vfr.v, (short)0, of[f], false, false);
      }
    }
    __syncthreads();
  }

  if (tid < 64) alpha[tid] = 1.f / lrow;
  __syncthreads();
  int rb = wave * 16 + ((lane >= 16) ? 8 : 0);
#pragma unroll
  for (int f = 0; f < 4; ++f)
#pragma unroll
    for (int r = 0; r < 8; ++r) {
      float val = of[f][r] * alpha[rb + r];
      yb[(size_t)(q0 + rb + r) * EMB + h * 64 + f * 16 + l15] = f2bf(val);
    }
}

// ---------------------------------------------------------------------------
extern "C" void kernel_launch(void* const* d_in, const int* in_sizes, int n_in,
                              void* d_out, int out_size, void* d_ws, size_t ws_size,
                              hipStream_t stream) {
  (void)in_sizes; (void)n_in; (void)out_size; (void)ws_size;
  const float* x      = (const float*)d_in[0];
  const float* w_attn = (const float*)d_in[1];
  const float* w_proj = (const float*)d_in[2];
  const float* qw     = (const float*)d_in[3];
  const float* kw     = (const float*)d_in[4];
  const float* vw     = (const float*)d_in[5];
  const float* cosb   = (const float*)d_in[6];
  const float* sinb   = (const float*)d_in[7];
  const float* beta   = (const float*)d_in[8];
  const float* g      = (const float*)d_in[9];
  float* out = (float*)d_out;

  char* ws = (char*)d_ws;
  size_t off = 0;
  auto alloc = [&](size_t bytes) -> void* {
    void* p = ws + off;
    off += (bytes + 255) & ~(size_t)255;
    return p;
  };
  unsigned short* xb   = (unsigned short*)alloc((size_t)T_LEN * EMB * 2);
  unsigned short* wabp = (unsigned short*)alloc((size_t)EMB * QKV3 * 2);  // packed
  unsigned short* wpbp = (unsigned short*)alloc((size_t)EMB * EMB * 2);   // packed
  float*          qkv  = (float*)alloc((size_t)T_LEN * QKV3 * 4);
  float*          qf   = (float*)alloc((size_t)T_LEN * EMB * 4);
  float*          kf   = (float*)alloc((size_t)T_LEN * EMB * 4);
  float*          vf   = (float*)alloc((size_t)T_LEN * EMB * 4);
  unsigned short* qbb  = (unsigned short*)alloc((size_t)T_LEN * EMB * 2);
  unsigned short* kbb  = (unsigned short*)alloc((size_t)T_LEN * EMB * 2);
  unsigned short* vob  = (unsigned short*)alloc((size_t)T_LEN * EMB * 2);
  unsigned short* ybb  = (unsigned short*)alloc((size_t)T_LEN * EMB * 2);

  cvt_f32_to_bf16<<<(T_LEN * EMB) / 256, 256, 0, stream>>>(x, xb, T_LEN * EMB);
  {
    int runs = (QKV3 / 16) * (EMB / 32) * 32;
    pack_b_bf16<<<(runs + 255) / 256, 256, 0, stream>>>(w_attn, wabp, EMB, QKV3);
  }
  {
    int runs = (EMB / 16) * (EMB / 32) * 32;
    pack_b_bf16<<<(runs + 255) / 256, 256, 0, stream>>>(w_proj, wpbp, EMB, EMB);
  }

  gemm_bf16_wmma<<<dim3(QKV3 / 128, T_LEN / 128), 256, 0, stream>>>(
      xb, wabp, qkv, T_LEN, QKV3, EMB);

  conv_silu_rope<<<T_LEN, 512, 0, stream>>>(qkv, qw, kw, vw, cosb, sinb,
                                            qf, kf, vf, qbb, kbb);

  delta_scan<<<NHEAD, 256, 0, stream>>>(qf, kf, vf, beta, g, vob);

  flash_attn<<<dim3(T_LEN / 64, NHEAD), 128, 0, stream>>>(qbb, kbb, vob, ybb);

  gemm_bf16_wmma<<<dim3(EMB / 128, T_LEN / 128), 256, 0, stream>>>(
      ybb, wpbp, out, T_LEN, EMB, EMB);
}
